// MultiHeadAttention_1915555414828
// MI455X (gfx1250) — matrix operand order repaired and checked by reading, not compiled, not run on hardware
//
#include <hip/hip_runtime.h>
#include <cstddef>

#define D_MODEL   1024
#define NUM_HEADS 16
#define D_K       64
#define BATCH     4
#define SEQ       2048
#define MTOK      (BATCH * SEQ)   // 8192 tokens

typedef __bf16 bf16_t;
typedef __attribute__((ext_vector_type(16))) __bf16 v16bf;
typedef __attribute__((ext_vector_type(8)))  __bf16 v8bf;
typedef __attribute__((ext_vector_type(4)))  __bf16 v4bf;
typedef __attribute__((ext_vector_type(8)))  float  v8f;
typedef __attribute__((ext_vector_type(4)))  float  v4f;
typedef __attribute__((ext_vector_type(4)))  unsigned int u32x4;
typedef __attribute__((ext_vector_type(8)))  int          i32x8;
typedef __attribute__((ext_vector_type(4)))  int          i32x4;

// LDS row pitch for the TDM-staged A panel: 1024 bf16 = 512 dwords per row,
// TDM pad inserts 4 dwords every 256 dwords -> 520 dwords = 2080 B pitch
// (16B-aligned; rows start at bank 8*r mod 64, only 2-way conflicted).
#define A_LDS_PITCH 2080

// ---------------- helpers ----------------

__device__ __forceinline__ bf16_t f2bf(float f) {
  union { float f; unsigned u; } x; x.f = f;
  unsigned r = x.u + 0x7FFFu + ((x.u >> 16) & 1u);   // round-to-nearest-even
  unsigned short h = (unsigned short)(r >> 16);
  union { unsigned short s; bf16_t b; } y; y.s = h;
  return y.b;
}

// A-operand (16x32 bf16, MxK): lane L holds row M=L&15,
// K = {(L>>4)*8 + 0..7} in comps 0..7 and {16 + (L>>4)*8 + 0..7} in comps 8..15.
// Source row-major with leading dim `ld` (elements). Two 16B loads per lane.
__device__ __forceinline__ v16bf load_a(const bf16_t* p, int ld, int lane) {
  const bf16_t* q = p + (size_t)(lane & 15) * ld + ((lane >> 4) << 3);
  v8bf lo = *(const v8bf*)q;
  v8bf hi = *(const v8bf*)(q + 16);
  v16bf r;
#pragma unroll
  for (int i = 0; i < 8; ++i) { r[i] = lo[i]; r[i + 8] = hi[i]; }
  return r;
}

// A-operand read from the TDM-padded LDS panel (pitch 2080 B, 16 B inserted
// after every 512 elements of a row). k is a multiple of 32 so the 32-col
// window never straddles a pad insert.
__device__ __forceinline__ v16bf load_a_lds(const unsigned char* base, int k, int lane) {
  const int r = lane & 15;
  const int c0 = k + ((lane >> 4) << 3);
  const unsigned char* p = base + r * A_LDS_PITCH + 2 * c0 + ((k >> 9) << 4);
  v8bf lo = *(const v8bf*)p;
  v8bf hi = *(const v8bf*)(p + 32);   // +16 elements
  v16bf v;
#pragma unroll
  for (int i = 0; i < 8; ++i) { v[i] = lo[i]; v[i + 8] = hi[i]; }
  return v;
}

// B-operand (32x16 bf16, KxN): lane L holds column N=L&15,
// K = {(L>>4)*8 + 0..7} in comps 0..7 and {16 + (L>>4)*8 + 0..7} in comps 8..15. Source is N-major (row n holds K contiguous).
__device__ __forceinline__ v16bf load_b(const bf16_t* p, int ld, int lane) {
  const bf16_t* q = p + (size_t)(lane & 15) * ld + ((lane >> 4) << 3);
  v8bf lo = *(const v8bf*)q;
  v8bf hi = *(const v8bf*)(q + 16);
  v16bf r;
#pragma unroll
  for (int i = 0; i < 8; ++i) { r[i] = lo[i]; r[i + 8] = hi[i]; }
  return r;
}

__device__ __forceinline__ v8f wmma_bf16(v16bf a, v16bf b, v8f c) {
  return __builtin_amdgcn_wmma_f32_16x16x32_bf16(
      /*neg_a=*/false, a, /*neg_b=*/false, b,
      /*c_mod=*/(short)0, c, /*reuse_a=*/false, /*reuse_b=*/false);
}

// TDM: DMA a 2-D bf16 tile [tile_d1 rows x tile_d0 cols] from global into LDS,
// with LDS padding of 4 dwords per 256 dwords (bank de-conflict).
// Descriptor layout per CDNA5 ISA (groups 0/1; 2-D so
// groups 2/3 are zero / unused).
__device__ __forceinline__ void tdm_load_tile_2d(
    unsigned lds_off, unsigned long long gaddr,
    unsigned tensor_d0, unsigned tensor_d1,
    unsigned tile_d0, unsigned tile_d1, unsigned stride_d0) {
  u32x4 g0;
  g0[0] = 1u;                                        // count=1, user descriptor
  g0[1] = lds_off;                                   // LDS byte offset
  g0[2] = (unsigned)(gaddr & 0xFFFFFFFFull);         // global_addr[31:0]
  g0[3] = (unsigned)((gaddr >> 32) & 0x01FFFFFFull)  // global_addr[56:32]
          | 0x80000000u;                             // type = 2 ("image")
  i32x8 g1;
  g1[0] = (int)((1u << 16)      // data_size = 2 bytes
              | (1u << 20)      // pad_enable
              | (7u << 22)      // pad_interval: 256 dwords
              | (3u << 25));    // pad_amount: 4 dwords (16 B)
  g1[1] = (int)(tensor_d0 << 16);                        // tensor_dim0 lo
  g1[2] = (int)((tensor_d0 >> 16) | (tensor_d1 << 16));  // dim0 hi | dim1 lo
  g1[3] = (int)((tensor_d1 >> 16) | (tile_d0 << 16));    // dim1 hi | tile_dim0
  g1[4] = (int)(tile_d1 & 0xFFFFu);                      // tile_dim1 (tile_dim2=0)
  g1[5] = (int)stride_d0;                                // tensor_dim0_stride lo
  g1[6] = 0;                                             // stride hi, dim1_stride lo
  g1[7] = 0;
  i32x4 z4; z4[0] = z4[1] = z4[2] = z4[3] = 0;
#if defined(__clang_major__) && __clang_major__ >= 23
  i32x8 z8;
#pragma unroll
  for (int i = 0; i < 8; ++i) z8[i] = 0;
  __builtin_amdgcn_tensor_load_to_lds(g0, g1, z4, z4, z8, 0);
#else
  __builtin_amdgcn_tensor_load_to_lds(g0, g1, z4, z4, 0);
#endif
}

// ---------------- f32 -> bf16 conversion ----------------

__global__ __launch_bounds__(256) void cvt_f32_bf16(
    const float* __restrict__ x, bf16_t* __restrict__ y, int n4) {
  int i = blockIdx.x * blockDim.x + threadIdx.x;
  if (i < n4) {
    v4f f = *(const v4f*)(x + (size_t)i * 4);
    v4bf o;
#pragma unroll
    for (int j = 0; j < 4; ++j) o[j] = f2bf(f[j]);
    *(v4bf*)(y + (size_t)i * 4) = o;
  }
}

// ---------------- projection GEMM: out = A @ W^T + bias ----------------
// The 16x1024 bf16 A panel for a block is DMA'd once into LDS by the Tensor
// Data Mover and shared by all 4 waves (each owning a 64-wide N slice);
// weights stream from global (L2-resident, reused by all 512 M-tiles).
// MODE 0: bf16 out, head-split [B,H,S,64]           (Q, K)
// MODE 1: bf16 out, head-split transposed [B,H,64,S] (V^T)
// MODE 2: f32 out, plain [M, 1024]                   (final projection)
template <int MODE>
__global__ __launch_bounds__(128) void gemm_wmma(
    const bf16_t* __restrict__ A, const bf16_t* __restrict__ W,
    const float* __restrict__ bias,
    bf16_t* __restrict__ dst_bf, float* __restrict__ dst_f) {
  __shared__ __align__(16) unsigned char a_lds[16 * A_LDS_PITCH];

  const int lane = threadIdx.x & 31;
  const int wave = threadIdx.x >> 5;
  const int m0 = blockIdx.x * 16;
  const int n0 = (blockIdx.y * 4 + wave) * 64;

  // One TDM descriptor per block: A[m0 .. m0+15][0 .. 1023] -> LDS (padded).
  if (wave == 0) {
    const unsigned lds_off = (unsigned)(size_t)(void*)a_lds;  // low 32b of flat = LDS offset
    const unsigned long long ga =
        (unsigned long long)(size_t)(const void*)(A + (size_t)m0 * D_MODEL);
    tdm_load_tile_2d(lds_off, ga, D_MODEL, MTOK, D_MODEL, 16, D_MODEL);
    __builtin_amdgcn_s_wait_tensorcnt(0);
  }
  __syncthreads();

  v8f acc[4];
#pragma unroll
  for (int j = 0; j < 4; ++j)
#pragma unroll
    for (int i = 0; i < 8; ++i) acc[j][i] = 0.f;

#pragma unroll 4
  for (int k = 0; k < D_MODEL; k += 32) {
    v16bf a = load_a_lds(a_lds, k, lane);
#pragma unroll
    for (int j = 0; j < 4; ++j) {
      v16bf b = load_b(W + (size_t)(n0 + j * 16) * D_MODEL + k, D_MODEL, lane);
      acc[j] = wmma_bf16(a, b, acc[j]);
    }
  }

  const int nlo = lane & 15;
  const int rb = (lane >> 4) * 8;   // C layout: comp i -> row rb+i, col nlo
#pragma unroll
  for (int j = 0; j < 4; ++j) {
    const int n = n0 + j * 16 + nlo;
    const float bv = bias[n];
#pragma unroll
    for (int i = 0; i < 8; ++i) {
      const int r = m0 + rb + i;
      const float val = acc[j][i] + bv;
      if (MODE == 2) {
        dst_f[(size_t)r * D_MODEL + n] = val;
      } else {
        const int bb = r >> 11;          // r / SEQ
        const int ss = r & (SEQ - 1);
        const int hh = n >> 6;
        const int dd = n & 63;
        const size_t idx = (MODE == 0)
            ? ((size_t)(bb * NUM_HEADS + hh) * SEQ + ss) * D_K + dd
            : ((size_t)(bb * NUM_HEADS + hh) * D_K + dd) * SEQ + ss;
        dst_bf[idx] = f2bf(val);
      }
    }
  }
}

// ---------------- causal flash attention ----------------
// One wave owns a 16-query tile; loops over 32-key chunks with online softmax.
// Q,K: [B,H,S,64] bf16.  Vt: [B,H,64,S] bf16 (pre-transposed so the PV
// B-operand is contiguous).  O: [B,S,H*64] bf16.
__global__ __launch_bounds__(128) void flash_attn(
    const bf16_t* __restrict__ Q, const bf16_t* __restrict__ K,
    const bf16_t* __restrict__ Vt, bf16_t* __restrict__ O) {
  __shared__ __align__(16) bf16_t p_lds[4][16][40];  // per-wave P staging, padded

  const int lane = threadIdx.x & 31;
  const int wave = threadIdx.x >> 5;
  const int q0 = (blockIdx.x * 4 + wave) * 16;
  const int h = blockIdx.y;
  const int b = blockIdx.z;
  const size_t bh = (size_t)(b * NUM_HEADS + h);

  const bf16_t* Qp = Q + (bh * SEQ + q0) * D_K;
  const v16bf a0 = load_a(Qp, D_K, lane);        // K-dim 0..31
  const v16bf a1 = load_a(Qp + 32, D_K, lane);   // K-dim 32..63

  const int nlo = lane & 15;
  const int rb = (lane >> 4) * 8;

  float rm[8], rl[8];
  v8f acc[4];
#pragma unroll
  for (int i = 0; i < 8; ++i) { rm[i] = -1e30f; rl[i] = 0.f; }
#pragma unroll
  for (int j = 0; j < 4; ++j)
#pragma unroll
    for (int i = 0; i < 8; ++i) acc[j][i] = 0.f;

  bf16_t (*pl)[40] = p_lds[wave];
  const int kmax = q0 + 16;   // causal: keys beyond the tile diagonal skipped

  for (int k0 = 0; k0 < kmax; k0 += 32) {
    const bf16_t* Kp = K + (bh * SEQ + k0) * D_K;
    const bf16_t* Vp = Vt + bh * D_K * (size_t)SEQ + k0;
    __builtin_prefetch(Kp + 32 * D_K, 0, 1);          // next key chunk
    __builtin_prefetch(Vp + 32, 0, 1);

    // S = Q @ K^T : two 16-key N-subtiles, K-dim 64 = 2 WMMA steps each
    v8f s[2];
#pragma unroll
    for (int t = 0; t < 2; ++t) {
      v16bf kb0 = load_b(Kp + t * 16 * D_K, D_K, lane);
      v16bf kb1 = load_b(Kp + t * 16 * D_K + 32, D_K, lane);
      v8f c;
#pragma unroll
      for (int i = 0; i < 8; ++i) c[i] = 0.f;
      c = wmma_bf16(a0, kb0, c);
      c = wmma_bf16(a1, kb1, c);
      s[t] = c;
    }

    // scale + causal mask (branch-free, per lane)
#pragma unroll
    for (int i = 0; i < 8; ++i) {
      const int qg = q0 + rb + i;
#pragma unroll
      for (int t = 0; t < 2; ++t) {
        const int kg = k0 + t * 16 + nlo;
        const float v = s[t][i] * 0.125f;          // 1/sqrt(d_k)
        s[t][i] = (kg <= qg) ? v : -1e30f;
      }
    }

    // online softmax: row stats live replicated across each 16-lane half
    float p0[8], p1[8];
#pragma unroll
    for (int i = 0; i < 8; ++i) {
      float m = fmaxf(s[0][i], s[1][i]);
#pragma unroll
      for (int o = 1; o < 16; o <<= 1) m = fmaxf(m, __shfl_xor(m, o, 32));
      const float mn = fmaxf(rm[i], m);
      const float so = __expf(rm[i] - mn);
      const float e0 = __expf(s[0][i] - mn);
      const float e1 = __expf(s[1][i] - mn);
      p0[i] = e0; p1[i] = e1;
      float ps = e0 + e1;
#pragma unroll
      for (int o = 1; o < 16; o <<= 1) ps += __shfl_xor(ps, o, 32);
      rl[i] = rl[i] * so + ps;
      rm[i] = mn;
#pragma unroll
      for (int j = 0; j < 4; ++j) acc[j][i] *= so;
    }

    // C-layout -> A-layout relayout of P through per-wave LDS.
    // Within-wave LDS ops are in-order on CDNA5; the explicit wait + memory
    // clobber stops the compiler from hoisting the loads above the stores.
#pragma unroll
    for (int i = 0; i < 8; ++i) {
      pl[rb + i][nlo]      = f2bf(p0[i]);
      pl[rb + i][16 + nlo] = f2bf(p1[i]);
    }
    asm volatile("s_wait_dscnt 0" ::: "memory");
    const v16bf pa = load_a(&pl[0][0], 40, lane);

    // O += P @ V  (4 d-subtiles of 16)
#pragma unroll
    for (int j = 0; j < 4; ++j) {
      v16bf vb = load_b(Vp + (size_t)(j * 16) * SEQ, SEQ, lane);
      acc[j] = wmma_bf16(pa, vb, acc[j]);
    }
  }

  // epilogue: normalize and scatter bf16 into [B,S,H*64]
#pragma unroll
  for (int j = 0; j < 4; ++j)
#pragma unroll
    for (int i = 0; i < 8; ++i) {
      const float o = acc[j][i] / rl[i];
      const int r = q0 + rb + i;
      O[((size_t)(b * SEQ + r)) * D_MODEL + h * D_K + j * 16 + nlo] = f2bf(o);
    }
}

// ---------------- host-side orchestration ----------------

extern "C" void kernel_launch(void* const* d_in, const int* in_sizes, int n_in,
                              void* d_out, int out_size, void* d_ws, size_t ws_size,
                              hipStream_t stream) {
  (void)in_sizes; (void)n_in; (void)out_size; (void)ws_size;

  const float* q   = (const float*)d_in[0];
  const float* k   = (const float*)d_in[1];
  const float* v   = (const float*)d_in[2];
  /* d_in[3] = mask: causal, applied analytically */
  const float* w_q = (const float*)d_in[4];
  const float* b_q = (const float*)d_in[5];
  const float* w_k = (const float*)d_in[6];
  const float* b_k = (const float*)d_in[7];
  const float* w_v = (const float*)d_in[8];
  const float* b_v = (const float*)d_in[9];
  const float* w_o = (const float*)d_in[10];
  const float* b_o = (const float*)d_in[11];
  float* out = (float*)d_out;

  char* ws = (char*)d_ws;
  size_t off = 0;
  auto alloc = [&](size_t bytes) -> char* {
    char* p = ws + off;
    off += (bytes + 255) & ~(size_t)255;
    return p;
  };

  const size_t tokBytes = (size_t)MTOK * D_MODEL * sizeof(bf16_t);      // 16 MB
  const size_t wBytes   = (size_t)D_MODEL * D_MODEL * sizeof(bf16_t);   // 2 MB

  bf16_t* qbf = (bf16_t*)alloc(tokBytes);
  bf16_t* kbf = (bf16_t*)alloc(tokBytes);
  bf16_t* vbf = (bf16_t*)alloc(tokBytes);
  bf16_t* wqb = (bf16_t*)alloc(wBytes);
  bf16_t* wkb = (bf16_t*)alloc(wBytes);
  bf16_t* wvb = (bf16_t*)alloc(wBytes);
  bf16_t* wob = (bf16_t*)alloc(wBytes);
  bf16_t* Qb  = (bf16_t*)alloc(tokBytes);   // [B,H,S,64]
  bf16_t* Kb  = (bf16_t*)alloc(tokBytes);   // [B,H,S,64]
  bf16_t* Vtb = (bf16_t*)alloc(tokBytes);   // [B,H,64,S]
  bf16_t* Ob  = (bf16_t*)alloc(tokBytes);   // [B,S,1024]

  // 1) f32 -> bf16 conversions
  {
    const int n4t = MTOK * D_MODEL / 4;
    const int n4w = D_MODEL * D_MODEL / 4;
    cvt_f32_bf16<<<(n4t + 255) / 256, 256, 0, stream>>>(q, qbf, n4t);
    cvt_f32_bf16<<<(n4t + 255) / 256, 256, 0, stream>>>(k, kbf, n4t);
    cvt_f32_bf16<<<(n4t + 255) / 256, 256, 0, stream>>>(v, vbf, n4t);
    cvt_f32_bf16<<<(n4w + 255) / 256, 256, 0, stream>>>(w_q, wqb, n4w);
    cvt_f32_bf16<<<(n4w + 255) / 256, 256, 0, stream>>>(w_k, wkb, n4w);
    cvt_f32_bf16<<<(n4w + 255) / 256, 256, 0, stream>>>(w_v, wvb, n4w);
    cvt_f32_bf16<<<(n4w + 255) / 256, 256, 0, stream>>>(w_o, wob, n4w);
  }

  // 2) QKV projections (TDM-staged A panel + WMMA), attention-friendly layouts
  const dim3 gg(MTOK / 16, (D_MODEL / 64) / 4);  // 512 x 4 blocks, 4 waves each
  gemm_wmma<0><<<gg, 128, 0, stream>>>(qbf, wqb, b_q, Qb,  nullptr);
  gemm_wmma<0><<<gg, 128, 0, stream>>>(kbf, wkb, b_k, Kb,  nullptr);
  gemm_wmma<1><<<gg, 128, 0, stream>>>(vbf, wvb, b_v, Vtb, nullptr);

  // 3) causal flash attention (WMMA + online softmax)
  flash_attn<<<dim3(SEQ / 64, NUM_HEADS, BATCH), 128, 0, stream>>>(Qb, Kb, Vtb, Ob);

  // 4) output projection -> f32 result
  gemm_wmma<2><<<gg, 128, 0, stream>>>(Ob, wob, b_o, nullptr, out);
}
